// PixelContrastLoss_9225589752668
// MI455X (gfx1250) — matrix operand order repaired and checked by reading, not compiled, not run on hardware
//
#include <hip/hip_runtime.h>
#include <hip/hip_bf16.h>
#include <math.h>

typedef __attribute__((ext_vector_type(16))) _Float16 v16h;
typedef __attribute__((ext_vector_type(8)))  float    v8f;

#define NVIEW   100
#define NROWS   200     // 2 views * 100 samples
#define PADROWS 208     // 13 * 16 (WMMA tile padding)
#define DDIM    128
#define NTILES  13

// ---------------------------------------------------------------------------
// Kernel 1: deterministic ordered selection. One wave32 per batch image.
// hard = first 50 indices with (label==0 && pred==0)  [matches torch semantics]
// easy = first 50 indices with (label==1 && pred==1)  [deterministic subset]
// Wave-ballot prefix sums keep slot assignment in strict index order.
// ---------------------------------------------------------------------------
__global__ void pcl_select(const int* __restrict__ labels,
                           const int* __restrict__ predict,
                           int* __restrict__ idx, int N) {
  const int b    = blockIdx.x;
  const int lane = threadIdx.x;                  // 32 threads = 1 wave
  const int* yh = labels  + (size_t)b * (size_t)N;
  const int* yp = predict + (size_t)b * (size_t)N;
  int* out = idx + b * NVIEW;
  const unsigned long long lanemask = (1ull << lane) - 1ull;

  int cntE = 0, cntH = 0;
  for (int base = 0; base < N && (cntE < 50 || cntH < 50); base += 32) {
    const int i = base + lane;
    const int l = yh[i];
    const int p = yp[i];
    const bool easy = (l == 1) && (p == 1);
    const bool hard = (l == 0) && (p == 0);
    const unsigned long long mE = __ballot(easy);
    const unsigned long long mH = __ballot(hard);
    if (easy) {
      const int slot = cntE + __popcll(mE & lanemask);
      if (slot < 50) out[slot] = i;              // X_[b, 0:50]  = easy
    }
    if (hard) {
      const int slot = cntH + __popcll(mH & lanemask);
      if (slot < 50) out[50 + slot] = i;         // X_[b, 50:100] = hard
    }
    cntE += __popcll(mE);
    cntH += __popcll(mH);
  }
  // Deterministically zero-fill if fewer than 50 matches exist.
  for (int t = (cntE < 50 ? cntE : 50) + lane; t < 50; t += 32) out[t] = 0;
  for (int t = (cntH < 50 ? cntH : 50) + lane; t < 50; t += 32) out[50 + t] = 0;
}

// ---------------------------------------------------------------------------
// Kernel 2: gather selected rows from feats [B, D, N] layout, L2-normalize,
// emit f16 matrix Ah[PADROWS][128]. Row r maps to (view s = r/2, batch b = r%2)
// per jnp.swapaxes(X_,0,1).reshape(200,D). Rows 200..207 are zero padding.
// ---------------------------------------------------------------------------
__global__ void pcl_gather_norm(const float* __restrict__ feats,
                                const int* __restrict__ idx,
                                _Float16* __restrict__ Ah, int N) {
  __shared__ float red[128];
  const int r = blockIdx.x;    // 0..207
  const int d = threadIdx.x;   // 0..127
  float v = 0.f;
  if (r < NROWS) {
    const int b = r & 1;
    const int s = r >> 1;
    const int n = idx[b * NVIEW + s];
    v = feats[((size_t)(b * DDIM + d)) * (size_t)N + (size_t)n];
  }
  red[d] = v * v;
  __syncthreads();
  for (int s2 = 64; s2 > 0; s2 >>= 1) {
    if (d < s2) red[d] += red[d + s2];
    __syncthreads();
  }
  const float inv = 1.f / fmaxf(sqrtf(red[0]), 1e-12f);   // F.normalize eps
  Ah[(size_t)r * DDIM + d] = (_Float16)(v * inv);
}

// ---------------------------------------------------------------------------
// Kernel 3: Gram matrix L = (A @ A^T) / BASE_TEMP via v_wmma_f32_16x16x32_f16.
// One wave32 per 16x16 output tile (EXEC all-ones as WMMA requires).
// A operand: lane = M row (lanes 0-15 & 16-31 both M=0..15), K split by lane
// half per ISA layout. B operand: lane = N column, K=0..7,16..23 on lanes 0-15,
// K=8..15,24..31 on lanes 16-31. C/D: VGPR v holds M = v + 8*(lane/16), N = lane%16.
// ---------------------------------------------------------------------------
__global__ void pcl_gram_wmma(const _Float16* __restrict__ Ah,
                              float* __restrict__ L) {
  const int tn   = blockIdx.x;
  const int tm   = blockIdx.y;
  const int lane = threadIdx.x;
  const int c16  = lane & 15;
  const int h    = lane >> 4;

  const _Float16* arow = Ah + (size_t)(tm * 16 + c16) * DDIM;
  const _Float16* brow = Ah + (size_t)(tn * 16 + c16) * DDIM;

  v8f acc = {};
#pragma unroll
  for (int k0 = 0; k0 < DDIM; k0 += 32) {
    v16h a, b;
#pragma unroll
    for (int i = 0; i < 8; ++i) {
      a[i]     = arow[k0 + 8 * h + i];        // VGPRs 0-3: K = 8h+0..7
      a[i + 8] = arow[k0 + 16 + 8 * h + i];   // VGPRs 4-7: K = 16+8h+0..7
    }
#pragma unroll
    for (int i = 0; i < 8; ++i) {
      b[i]     = brow[k0 + 8 * h + i];        // K = 8h+0..7, N = c16
      b[i + 8] = brow[k0 + 16 + 8 * h + i];   // K = 16+8h+0..7
    }
    acc = __builtin_amdgcn_wmma_f32_16x16x32_f16(
        /*neg_a=*/false, a, /*neg_b=*/false, b,
        /*c_mod=*/(short)0, acc, /*reuse_a=*/false, /*reuse_b=*/false);
  }
#pragma unroll
  for (int v = 0; v < 8; ++v) {
    const int row = tm * 16 + v + 8 * h;
    const int col = tn * 16 + c16;
    L[(size_t)row * PADROWS + col] = acc[v] * 10.0f;   // 1/BASE_TEMP = 10
  }
}

// ---------------------------------------------------------------------------
// Kernel 4: per-row masked softmax statistics.
//   m   = max_j L[i][j]
//   neg = sum_{j: different 100-block} exp(L-m)
//   lp  = sum over mask (same block, minus diagonal for i<100) of
//         (L-m) - log(exp(L-m) + neg)
//   mean_lpp[i] = lp / (i<100 ? 99 : 100)
// ---------------------------------------------------------------------------
__global__ void pcl_row_loss(const float* __restrict__ L,
                             float* __restrict__ mlpp) {
  __shared__ float red[256];
  const int i = blockIdx.x;
  const int j = threadIdx.x;
  const float l = (j < NROWS) ? L[(size_t)i * PADROWS + j] : -INFINITY;

  red[j] = l;
  __syncthreads();
  for (int s = 128; s > 0; s >>= 1) {
    if (j < s) red[j] = fmaxf(red[j], red[j + s]);
    __syncthreads();
  }
  const float m = red[0];
  __syncthreads();

  const float e = (j < NROWS) ? expf(l - m) : 0.f;
  const bool sameBlk = (j / NVIEW) == (i / NVIEW);

  red[j] = (j < NROWS && !sameBlk) ? e : 0.f;     // neg_mask = 1 - mask0
  __syncthreads();
  for (int s = 128; s > 0; s >>= 1) {
    if (j < s) red[j] += red[j + s];
    __syncthreads();
  }
  const float neg = red[0];
  __syncthreads();

  const bool maskv = (j < NROWS) && sameBlk && !((i < NVIEW) && (j == i));
  red[j] = maskv ? ((l - m) - logf(e + neg)) : 0.f;
  __syncthreads();
  for (int s = 128; s > 0; s >>= 1) {
    if (j < s) red[j] += red[j + s];
    __syncthreads();
  }
  if (j == 0) mlpp[i] = red[0] / ((i < NVIEW) ? 99.f : 100.f);
}

// ---------------------------------------------------------------------------
// Kernel 5: loss = -(mean(mean_lpp) / TEMP),  TEMP = 20
// ---------------------------------------------------------------------------
__global__ void pcl_final(const float* __restrict__ mlpp,
                          float* __restrict__ out) {
  __shared__ float red[256];
  const int j = threadIdx.x;
  red[j] = (j < NROWS) ? mlpp[j] : 0.f;
  __syncthreads();
  for (int s = 128; s > 0; s >>= 1) {
    if (j < s) red[j] += red[j + s];
    __syncthreads();
  }
  if (j == 0) out[0] = -(red[0] / (float)NROWS) / 20.0f;
}

// ---------------------------------------------------------------------------
extern "C" void kernel_launch(void* const* d_in, const int* in_sizes, int n_in,
                              void* d_out, int out_size, void* d_ws, size_t ws_size,
                              hipStream_t stream) {
  (void)n_in; (void)out_size; (void)ws_size;
  const float* feats   = (const float*)d_in[0];   // [2, 128, 32, 64, 64]
  const int*   labels  = (const int*)d_in[1];     // [2, 32, 64, 64]
  const int*   predict = (const int*)d_in[2];     // [2, 32, 64, 64]
  float*       out     = (float*)d_out;           // scalar f32 loss

  const int N = in_sizes[1] / 2;                  // flattened spatial = 131072

  // Workspace layout (all offsets 256B-aligned):
  char* ws = (char*)d_ws;
  int*       idx  = (int*)(ws + 0);               //   800 B  (2 x 100 ints)
  _Float16*  Ah   = (_Float16*)(ws + 1024);       // 53248 B  (208 x 128 f16)
  float*     L    = (float*)(ws + 54528);         // 173056 B (208 x 208 f32)
  float*     mlpp = (float*)(ws + 227840);        //   800 B  (200 f32)

  pcl_select<<<2, 32, 0, stream>>>(labels, predict, idx, N);
  pcl_gather_norm<<<PADROWS, 128, 0, stream>>>(feats, idx, Ah, N);
  pcl_gram_wmma<<<dim3(NTILES, NTILES), 32, 0, stream>>>(Ah, L);
  pcl_row_loss<<<NROWS, 256, 0, stream>>>(L, mlpp);
  pcl_final<<<1, 256, 0, stream>>>(mlpp, out);
}
